// S6Block_37529424232986
// MI455X (gfx1250) — matrix operand order repaired and checked by reading, not compiled, not run on hardware
//
#include <hip/hip_runtime.h>
#include <hip/hip_bf16.h>
#include <math.h>

// Problem constants (match reference)
#define B_   4
#define L_   4096
#define D_   64
#define N_   16
#define DT_  0.1f
#define EPS_ 1e-6f

// Chunked-scan configuration: 16 chunks x 4096 channels = 65536 threads = 2048 wave32s
#define NCHUNK 16
#define CL     (L_ / NCHUNK)    // 256 steps per chunk
#define CH     (B_ * D_ * N_)   // 4096 channels
#define PFD    16               // prefetch distance (iterations)

typedef __attribute__((ext_vector_type(16))) _Float16 v16h;
typedef __attribute__((ext_vector_type(8)))  float    v8f;

// ---------------------------------------------------------------------------
// B-matrix fragment (K x N = 32 x 16, f16) per ISA 7.12.2:
//   lanes 0-15 : N = lane,    v in 0..3: K = 2v..2v+1, v in 4..7: K = 16+2(v-4)..+1
//   lanes 16-31: N = lane-16, v in 0..3: K = 8+2v..+1, v in 4..7: K = 24+2(v-4)..+1
// W is (N_ x D_) row-major; B[k][n] = W[n][k]; ks selects K-step (0: K=0..31, 1: K=32..63)
// ---------------------------------------------------------------------------
static __device__ __forceinline__ v16h load_b_frag(const float* __restrict__ W,
                                                   int ncol, int half, int ks) {
  v16h f;
#pragma unroll
  for (int v = 0; v < 8; ++v) {
    const int k = ks * 32 + ((v < 4) ? (half * 8 + 2 * v) : (16 + half * 8 + 2 * (v - 4)));
    f[2 * v]     = (_Float16)W[ncol * D_ + k];
    f[2 * v + 1] = (_Float16)W[ncol * D_ + k + 1];
  }
  return f;
}

// ---------------------------------------------------------------------------
// Kernel 1: projections via WMMA  (Bm = x Wb^T + bb, Cm = x Wc^T + bc)
// plus the delta network (softplus(x Wd^T + bd) + DT).
// One wave per 16-token tile; K = 64 = two 16x16x32 WMMA k-steps per output.
// ---------------------------------------------------------------------------
__global__ __launch_bounds__(128) void s6_proj_wmma(
    const float* __restrict__ x,
    const float* __restrict__ Wb, const float* __restrict__ bb,
    const float* __restrict__ Wc, const float* __restrict__ bc,
    const float* __restrict__ Wd, const float* __restrict__ bd,
    float* __restrict__ Bm, float* __restrict__ Cm, float* __restrict__ dlt)
{
  const int gtid = blockIdx.x * blockDim.x + threadIdx.x;
  const int lane = gtid & 31;
  const int tile = gtid >> 5;            // 0 .. (B_*L_/16 - 1) = 0..1023
  const int row0 = tile * 16;            // flattened (b*L + l) row base
  const int mrow = lane & 15;
  const int half = lane >> 4;

  // A-matrix fragments (M x K = 16 x 32, f16), ISA 7.12.2 layout:
  //   v in 0..3: K = half*8 + 2v ;  v in 4..7: K = 16 + half*8 + 2(v-4)
  v16h a0, a1;
  const float* xr = x + (size_t)(row0 + mrow) * D_;
#pragma unroll
  for (int v = 0; v < 8; ++v) {
    const int kb = (v < 4) ? (half * 8 + 2 * v) : (16 + half * 8 + 2 * (v - 4));
    a0[2 * v]     = (_Float16)xr[kb];
    a0[2 * v + 1] = (_Float16)xr[kb + 1];
    a1[2 * v]     = (_Float16)xr[32 + kb];
    a1[2 * v + 1] = (_Float16)xr[32 + kb + 1];
  }

  const int ncol = lane & 15;

  // Bm accumulation: two k-steps
  v8f accB = {};
  {
    v16h f0 = load_b_frag(Wb, ncol, half, 0);
    accB = __builtin_amdgcn_wmma_f32_16x16x32_f16(false, a0, false, f0,
                                                  (short)0, accB, false, false);
    v16h f1 = load_b_frag(Wb, ncol, half, 1);
    accB = __builtin_amdgcn_wmma_f32_16x16x32_f16(false, a1, false, f1,
                                                  (short)0, accB, false, false);
  }
  // Cm accumulation
  v8f accC = {};
  {
    v16h f0 = load_b_frag(Wc, ncol, half, 0);
    accC = __builtin_amdgcn_wmma_f32_16x16x32_f16(false, a0, false, f0,
                                                  (short)0, accC, false, false);
    v16h f1 = load_b_frag(Wc, ncol, half, 1);
    accC = __builtin_amdgcn_wmma_f32_16x16x32_f16(false, a1, false, f1,
                                                  (short)0, accC, false, false);
  }

  // Store: C/D layout VGPR r -> M = half*8 + r, N = ncol
  const float biasB = bb[ncol];
  const float biasC = bc[ncol];
#pragma unroll
  for (int r = 0; r < 8; ++r) {
    const int m = half * 8 + r;
    Bm[(size_t)(row0 + m) * N_ + ncol] = accB[r] + biasB;
    Cm[(size_t)(row0 + m) * N_ + ncol] = accC[r] + biasC;
  }

  // Delta network: 16 rows per wave, lanes 0..15 each own one row (after WMMA,
  // so EXEC was all-ones for the matrix ops).
  if (lane < 16) {
    const int row = row0 + lane;
    const float* xr2 = x + (size_t)row * D_;
    float acc = bd[0];
#pragma unroll
    for (int k = 0; k < D_; ++k) acc = fmaf(xr2[k], Wd[k], acc);
    const float sp = (acc > 20.f) ? acc : log1pf(__expf(acc));   // softplus
    dlt[row] = sp + DT_;
  }
}

// index decode helper: g = chunk*CH + ((b*D_ + d)*N_ + n)
#define DECODE_G()                         \
  const int n     = g & 15;                \
  const int d     = (g >> 4) & 63;         \
  const int b     = (g >> 10) & 3;         \
  const int chunk = g >> 12;

// ---------------------------------------------------------------------------
// Kernel 2 (phase A): per-chunk local scan -> (P = prod Abar, S = local h_end)
// ---------------------------------------------------------------------------
__global__ __launch_bounds__(256) void s6_scan_chunk(
    const float* __restrict__ x, const float* __restrict__ A,
    const float* __restrict__ Bm, const float* __restrict__ dlt,
    float* __restrict__ Pbuf, float* __restrict__ Sbuf)
{
  const int g = blockIdx.x * blockDim.x + threadIdx.x;   // 0 .. NCHUNK*CH-1
  DECODE_G();
  const float Av = A[d * N_ + n];                        // = -(n+1)
  const int l0 = chunk * CL;

  const float* dtp = dlt + (size_t)b * L_;
  const float* bmp = Bm + ((size_t)b * L_) * N_ + n;
  const float* xp  = x  + ((size_t)b * L_) * D_ + d;

  float h = 0.f, P = 1.f;
  for (int i = 0; i < CL; ++i) {
    const int l = l0 + i;
    if (i + PFD < CL) {                                  // uniform per wave
      __builtin_prefetch(xp  + (size_t)(l + PFD) * D_, 0, 1);
      __builtin_prefetch(bmp + (size_t)(l + PFD) * N_, 0, 1);
    }
    const float dt   = dtp[l];
    const float tmp  = Av * dt;
    const float abar = __expf(tmp);
    const float bm   = bmp[(size_t)l * N_];
    const float xv   = xp[(size_t)l * D_];
    const float bbar = (abar - 1.f) * (dt * bm) / (tmp + EPS_);
    h = fmaf(abar, h, bbar * xv);
    P *= abar;
  }
  Pbuf[g] = P;
  Sbuf[g] = h;
}

// ---------------------------------------------------------------------------
// Kernel 3 (phase B): sequential combine over the 16 chunks -> carry-in per chunk
// ---------------------------------------------------------------------------
__global__ __launch_bounds__(256) void s6_scan_combine(
    const float* __restrict__ Pbuf, const float* __restrict__ Sbuf,
    float* __restrict__ Hin)
{
  const int c = blockIdx.x * blockDim.x + threadIdx.x;   // 0 .. CH-1
  float h = 0.f;
#pragma unroll
  for (int k = 0; k < NCHUNK; ++k) {
    const int idx = k * CH + c;
    Hin[idx] = h;
    h = fmaf(Pbuf[idx], h, Sbuf[idx]);
  }
}

// ---------------------------------------------------------------------------
// Kernel 4 (phase C): re-run local scan with carry-in, reduce over n, emit out.
// Wave layout: lanes 0-15 = n for d0, lanes 16-31 = n for d1 -> 16-wide shuffles.
// ---------------------------------------------------------------------------
__global__ __launch_bounds__(256) void s6_scan_emit(
    const float* __restrict__ x, const float* __restrict__ A,
    const float* __restrict__ Bm, const float* __restrict__ Cm,
    const float* __restrict__ dlt, const float* __restrict__ Hin,
    float* __restrict__ out)
{
  const int g = blockIdx.x * blockDim.x + threadIdx.x;
  DECODE_G();
  const float Av = A[d * N_ + n];
  const int l0 = chunk * CL;

  const float* dtp = dlt + (size_t)b * L_;
  const float* bmp = Bm + ((size_t)b * L_) * N_ + n;
  const float* cmp = Cm + ((size_t)b * L_) * N_ + n;
  const float* xp  = x  + ((size_t)b * L_) * D_ + d;

  float h = Hin[g];
  for (int i = 0; i < CL; ++i) {
    const int l = l0 + i;
    if (i + PFD < CL) {
      __builtin_prefetch(xp  + (size_t)(l + PFD) * D_, 0, 1);
      __builtin_prefetch(bmp + (size_t)(l + PFD) * N_, 0, 1);
      __builtin_prefetch(cmp + (size_t)(l + PFD) * N_, 0, 1);
    }
    const float dt   = dtp[l];
    const float tmp  = Av * dt;
    const float abar = __expf(tmp);
    const float bm   = bmp[(size_t)l * N_];
    const float xv   = xp[(size_t)l * D_];
    const float bbar = (abar - 1.f) * (dt * bm) / (tmp + EPS_);
    h = fmaf(abar, h, bbar * xv);

    // out[b,l,d] = sum_n h * Cm[b,l,n] : reduce over the 16-lane n-group
    float val = h * cmp[(size_t)l * N_];
    val += __shfl_xor(val, 8, 16);
    val += __shfl_xor(val, 4, 16);
    val += __shfl_xor(val, 2, 16);
    val += __shfl_xor(val, 1, 16);
    if (n == 0) out[((size_t)b * L_ + l) * D_ + d] = val;
  }
}

// ---------------------------------------------------------------------------
// Launch: proj (WMMA) -> chunk scan -> combine -> emit, all on `stream`.
// Workspace layout (floats): Bm | Cm | dlt | Pbuf | Sbuf | Hin  (~2.95 MB)
// ---------------------------------------------------------------------------
extern "C" void kernel_launch(void* const* d_in, const int* in_sizes, int n_in,
                              void* d_out, int out_size, void* d_ws, size_t ws_size,
                              hipStream_t stream) {
  const float* x  = (const float*)d_in[0];
  const float* A  = (const float*)d_in[1];
  const float* Wb = (const float*)d_in[2];
  const float* bb = (const float*)d_in[3];
  const float* Wc = (const float*)d_in[4];
  const float* bc = (const float*)d_in[5];
  const float* Wd = (const float*)d_in[6];
  const float* bd = (const float*)d_in[7];
  float* out = (float*)d_out;

  float* ws   = (float*)d_ws;
  float* Bm   = ws;                                   // B_*L_*N_  = 262144
  float* Cm   = Bm  + (size_t)B_ * L_ * N_;           // 262144
  float* dlt  = Cm  + (size_t)B_ * L_ * N_;           // B_*L_     = 16384
  float* Pbuf = dlt + (size_t)B_ * L_;                // NCHUNK*CH = 65536
  float* Sbuf = Pbuf + (size_t)NCHUNK * CH;           // 65536
  float* Hin  = Sbuf + (size_t)NCHUNK * CH;           // 65536

  // 1024 tiles of 16 tokens, one wave each -> 256 blocks x 128 threads
  s6_proj_wmma<<<256, 128, 0, stream>>>(x, Wb, bb, Wc, bc, Wd, bd, Bm, Cm, dlt);
  // 65536 threads = 2048 wave32s
  s6_scan_chunk<<<(NCHUNK * CH) / 256, 256, 0, stream>>>(x, A, Bm, dlt, Pbuf, Sbuf);
  s6_scan_combine<<<CH / 256, 256, 0, stream>>>(Pbuf, Sbuf, Hin);
  s6_scan_emit<<<(NCHUNK * CH) / 256, 256, 0, stream>>>(x, A, Bm, Cm, dlt, Hin, out);
}
